// SuperBox_22368189678244
// MI455X (gfx1250) — matrix operand order repaired and checked by reading, not compiled, not run on hardware
//
#include <hip/hip_runtime.h>
#include <hip/hip_bf16.h>

// ---------------------------------------------------------------------------
// Problem constants (from the reference)
// ---------------------------------------------------------------------------
constexpr int cA  = 8;       // batch (agents)
constexpr int cK  = 50;      // boxes per agent
constexpr int cH  = 100;
constexpr int cW  = 352;
constexpr int cHW = cH * cW; // 35200, divisible by 128

// ---------------------------------------------------------------------------
// CDNA5 WMMA types
// ---------------------------------------------------------------------------
typedef __attribute__((ext_vector_type(16))) __bf16        v16bf;
typedef __attribute__((ext_vector_type(8)))  float         v8f;
typedef __attribute__((ext_vector_type(4)))  unsigned int  u32x4;

union Frag {
    v16bf        bf;
    u32x4        q[2];
    unsigned int u[8];
};

__device__ __forceinline__ unsigned short f2bf(float f) {
    unsigned int u = __float_as_uint(f);
    u += 0x7FFFu + ((u >> 16) & 1u);   // round-to-nearest-even to bf16
    return (unsigned short)(u >> 16);
}
__device__ __forceinline__ float bf2f(unsigned short h) {
    return __uint_as_float(((unsigned int)h) << 16);
}

// ---------------------------------------------------------------------------
// Layout transforms (run once per call, tiny cost)
// ---------------------------------------------------------------------------
// feature [A, C=64, H, W] f32  ->  [A, H, W, 64] bf16
__global__ __launch_bounds__(256) void feat_to_nhwc_bf16(
    const float* __restrict__ f, unsigned short* __restrict__ out) {
    int n = blockIdx.x * 256 + threadIdx.x;
    int total = cA * 64 * cHW;
    if (n >= total) return;
    int a   = n / (64 * cHW);
    int rem = n - a * (64 * cHW);
    int c   = rem / cHW;
    int p   = rem - c * cHW;
    out[((size_t)a * cHW + p) * 64 + c] = f2bf(f[n]);
}

// w [O, I, KH, KW] f32  ->  wt [O, KH*KW, I] bf16   (K ordering: tap-major, ci-minor)
__global__ __launch_bounds__(256) void weight_to_bf16(
    const float* __restrict__ w, unsigned short* __restrict__ wt,
    int O, int I, int KK) {
    int n = blockIdx.x * 256 + threadIdx.x;
    int total = O * I * KK;
    if (n >= total) return;
    int o   = n / (I * KK);
    int rem = n - o * (I * KK);
    int i   = rem / KK;
    int kp  = rem - i * KK;
    wt[((size_t)o * KK + kp) * I + i] = f2bf(w[n]);
}

// ---------------------------------------------------------------------------
// Implicit-GEMM conv + bias + ReLU, bf16 in/out, f32 WMMA accumulate.
//   in  : [A, H, W, CIN]   bf16
//   wt  : [Cout, KH*KH, CIN] bf16
//   out : [A, H, W, Cout]  bf16
// GEMM: M = Cout (A = weights), N = spatial (B = patches), K = KH*KH*CIN in
// steps of 32 via V_WMMA_F32_16X16X32_BF16.
// Block: 256 threads = 8 wave32s; workgroup tile = 64 couts x 128 positions.
//   wave w: cout fragment (w&3)*16, position slab (w>>2)*64 = 4 x 16-pos
//   subtiles -> 4 accumulators sharing one A fragment (A-reuse = 4).
// Compile-time CIN/KH: channel loop fully unrolls, no integer division.
// ---------------------------------------------------------------------------
template <int CIN, int KH>
__global__ __launch_bounds__(256) void conv_wmma_bf16(
    const unsigned short* __restrict__ in,
    const unsigned short* __restrict__ wt,
    const float*          __restrict__ bias,
    unsigned short*       __restrict__ out,
    int Cout) {

    constexpr int PAD  = (KH == 3) ? 1 : 0;
    constexpr int KK   = KH * KH;
    constexpr int Ktot = KK * CIN;

    const int tid  = threadIdx.x;
    const int lane = tid & 31;
    const int wave = tid >> 5;
    const int l16  = lane & 15;
    const int hi   = lane >> 4;          // lane half selects K sub-range
    const int mI   = wave & 3;           // cout 16-group within 64
    const int nw   = wave >> 2;          // position 64-slab within 128

    const int a       = blockIdx.z;
    const int co_frag = blockIdx.y * 64 + mI * 16;
    const int pbase   = blockIdx.x * 128 + nw * 64;

    // B-fragment columns: lane l16 owns one spatial position per subtile
    int py[4], px[4];
    #pragma unroll
    for (int s = 0; s < 4; ++s) {
        int p = pbase + s * 16 + l16;
        py[s] = p / cW;
        px[s] = p - py[s] * cW;
    }

    v8f acc[4] = {{}, {}, {}, {}};

    // A fragment: row M = co_frag + l16; contiguous K {hi*16 .. hi*16+15}
    const unsigned short* wrow = wt + (size_t)(co_frag + l16) * Ktot + hi * 16;

    #pragma unroll 1
    for (int kp = 0; kp < KK; ++kp) {
        const int ky = kp / KH;              // KH is constexpr -> cheap
        const int dy = ky - PAD;
        const int dx = (kp - ky * KH) - PAD;

        const unsigned short* ir[4];
        #pragma unroll
        for (int s = 0; s < 4; ++s) {
            const int yy = py[s] + dy, xx = px[s] + dx;
            const bool ok = ((unsigned)yy < (unsigned)cH) &
                            ((unsigned)xx < (unsigned)cW);
            // B fragment: col N = l16, contiguous K {hi*16 .. hi*16+15}
            ir[s] = ok ? in + ((size_t)(a * cH + yy) * cW + xx) * CIN + hi * 16
                       : nullptr;
        }
        const unsigned short* wkp = wrow + kp * CIN;
        if (kp + 1 < KK) __builtin_prefetch(wkp + CIN, 0, 1);  // next tap's weights

        #pragma unroll
        for (int c0 = 0; c0 < CIN; c0 += 32) {
            Frag Af;
            Af.q[0] = *(const u32x4*)(wkp + c0);          // K hi*16 + 0..7
            Af.q[1] = *(const u32x4*)(wkp + c0 + 8);      // K hi*16 + 8..15
            #pragma unroll
            for (int s = 0; s < 4; ++s) {
                Frag B;
                if (ir[s]) {
                    B.q[0] = *(const u32x4*)(ir[s] + c0);      // K hi*16 + 0..7
                    B.q[1] = *(const u32x4*)(ir[s] + c0 + 8);  // K hi*16 + 8..15
                } else {
                    #pragma unroll
                    for (int i = 0; i < 8; ++i) B.u[i] = 0u;
                }
                acc[s] = __builtin_amdgcn_wmma_f32_16x16x32_bf16(
                             false, Af.bf, false, B.bf, (short)0, acc[s],
                             false, false);
            }
        }
    }

    // D layout: VGPR v, lane (l16,hi) -> (M = v + 8*hi, N = l16)
    // => each lane holds 8 contiguous couts [co_frag+hi*8 .. +7] for one pixel.
    const float* bp = bias + co_frag + hi * 8;
    float bl[8];
    #pragma unroll
    for (int v = 0; v < 8; ++v) bl[v] = bp[v];

    #pragma unroll
    for (int s = 0; s < 4; ++s) {
        union Opack { unsigned short h[8]; u32x4 q; } o;
        #pragma unroll
        for (int v = 0; v < 8; ++v) {
            float f = acc[s][v] + bl[v];
            f = f > 0.f ? f : 0.f;
            o.h[v] = f2bf(f);
        }
        const int p = pbase + s * 16 + l16;
        const size_t po = ((size_t)a * cHW + p) * Cout + co_frag + hi * 8;
        *(u32x4*)(out + po) = o.q;   // 16B store: 8 couts, one pixel
    }
}

// ---------------------------------------------------------------------------
// Box-mean pooling over the final [A,H,W,256] bf16 map.
// One block per (a,k) box, one thread per channel; coalesced channel reads.
// Equivalent to the reference's integral-image gather (sum over [u,d) x [l,r)).
// ---------------------------------------------------------------------------
__global__ __launch_bounds__(256) void box_pool(
    const unsigned short* __restrict__ act,
    const int* __restrict__ boxes,
    float* __restrict__ out) {
    const int ak = blockIdx.x;           // a*K + k
    const int a  = ak / cK;
    const int c  = threadIdx.x;          // channel 0..255
    const int* bx = boxes + (size_t)ak * 4;
    const int l = bx[0], r = bx[1], u = bx[2], d = bx[3];
    float acc = 0.f;
    for (int yy = u; yy < d; ++yy) {
        const unsigned short* row = act + ((size_t)(a * cH + yy) * cW + l) * 256 + c;
        for (int xx = l; xx < r; ++xx) {
            acc += bf2f(*row);
            row += 256;
        }
    }
    const float area = (float)((d - u) * (r - l));
    out[(size_t)ak * 256 + c] = acc / area;
}

// ---------------------------------------------------------------------------
// Host driver
// ---------------------------------------------------------------------------
extern "C" void kernel_launch(void* const* d_in, const int* in_sizes, int n_in,
                              void* d_out, int out_size, void* d_ws, size_t ws_size,
                              hipStream_t stream) {
    (void)in_sizes; (void)n_in; (void)out_size; (void)ws_size;

    const float* feature = (const float*)d_in[0];
    const int*   boxes   = (const int*)d_in[1];

    struct Layer { const float* w; const float* b; int O, I, KH; };
    Layer Ls[9] = {
        { (const float*)d_in[ 2], (const float*)d_in[ 3], 128,  64, 3 },  // w1a
        { (const float*)d_in[ 4], (const float*)d_in[ 5], 128, 128, 3 },  // w1b
        { (const float*)d_in[ 6], (const float*)d_in[ 7], 128, 128, 3 },  // w2a
        { (const float*)d_in[ 8], (const float*)d_in[ 9], 128, 128, 3 },  // w2b
        { (const float*)d_in[10], (const float*)d_in[11], 256, 128, 3 },  // w3a
        { (const float*)d_in[12], (const float*)d_in[13], 256, 256, 3 },  // w3b
        { (const float*)d_in[14], (const float*)d_in[15], 256, 256, 3 },  // wDa
        { (const float*)d_in[16], (const float*)d_in[17], 256, 256, 1 },  // wDb (1x1)
        { (const float*)d_in[18], (const float*)d_in[19], 256, 256, 1 },  // wF  (1x1)
    };

    // --- workspace carve (256B aligned) ---
    char*  ws  = (char*)d_ws;
    size_t off = 0;
    auto carve = [&](size_t bytes) -> char* {
        off = (off + 255) & ~(size_t)255;
        char* p = ws + off;
        off += bytes;
        return p;
    };
    unsigned short* act0 = (unsigned short*)carve((size_t)cA * cHW *  64 * 2);
    unsigned short* actA = (unsigned short*)carve((size_t)cA * cHW * 256 * 2);
    unsigned short* actB = (unsigned short*)carve((size_t)cA * cHW * 256 * 2);
    unsigned short* wts[9];
    for (int i = 0; i < 9; ++i)
        wts[i] = (unsigned short*)carve((size_t)Ls[i].O * Ls[i].I * Ls[i].KH * Ls[i].KH * 2);

    // --- layout transforms ---
    {
        int total = cA * 64 * cHW;
        feat_to_nhwc_bf16<<<(total + 255) / 256, 256, 0, stream>>>(feature, act0);
    }
    for (int i = 0; i < 9; ++i) {
        int KK    = Ls[i].KH * Ls[i].KH;
        int total = Ls[i].O * Ls[i].I * KK;
        weight_to_bf16<<<(total + 255) / 256, 256, 0, stream>>>(
            Ls[i].w, wts[i], Ls[i].O, Ls[i].I, KK);
    }

    // --- conv stack (ping-pong) ---
    const unsigned short* cur = act0;
    unsigned short*       nxt = actA;
    for (int i = 0; i < 9; ++i) {
        dim3 grid(cHW / 128, Ls[i].O / 64, cA);
        if (Ls[i].KH == 3 && Ls[i].I == 64) {
            conv_wmma_bf16< 64, 3><<<grid, 256, 0, stream>>>(cur, wts[i], Ls[i].b, nxt, Ls[i].O);
        } else if (Ls[i].KH == 3 && Ls[i].I == 128) {
            conv_wmma_bf16<128, 3><<<grid, 256, 0, stream>>>(cur, wts[i], Ls[i].b, nxt, Ls[i].O);
        } else if (Ls[i].KH == 3 && Ls[i].I == 256) {
            conv_wmma_bf16<256, 3><<<grid, 256, 0, stream>>>(cur, wts[i], Ls[i].b, nxt, Ls[i].O);
        } else {
            conv_wmma_bf16<256, 1><<<grid, 256, 0, stream>>>(cur, wts[i], Ls[i].b, nxt, Ls[i].O);
        }
        cur = nxt;
        nxt = (cur == actA) ? actB : actA;
    }
    // 9 hops starting act0->actA leave the final map in actA (== cur).

    // --- box-mean pooling -> d_out [A,K,256] f32 ---
    box_pool<<<cA * cK, 256, 0, stream>>>(cur, boxes, (float*)d_out);
}
